// LongformerResidual_16174846837343
// MI455X (gfx1250) — matrix operand order repaired and checked by reading, not compiled, not run on hardware
//
#include <hip/hip_runtime.h>
#include <hip/hip_bf16.h>
#include <math.h>

// ---------------------------------------------------------------------------
// Longformer forward for MI455X (gfx1250, wave32, WMMA + async global->LDS).
// GEMMs: f16 A (activations) x f16 B^T (pre-transposed weights), f32 acc.
// Inner loop: double-buffered global_load_async_to_lds_b128 (ASYNCcnt),
// one barrier per K-step, 4 WMMAs per wave per K-step (32x32 wave tile).
// Attention: flash-style sliding window, WMMA for Q.K^T and P.V.
// ---------------------------------------------------------------------------

typedef __attribute__((ext_vector_type(16))) _Float16 v16h;
typedef __attribute__((ext_vector_type(8)))  float    v8f;

#define S_LEN   4096
#define D_MODEL 768
#define NHEAD   12
#define DHEAD   64
#define FFDIM   3072
#define NLAYER  12
#define WWIN    256
#define NEGBIG  -1.0e9f

// ---------------- CDNA5 async global->LDS copy (16 bytes per lane) ----------
static __device__ __forceinline__ unsigned int lds_off_of(const void* p) {
  // flat shared address: low 32 bits are the wave-relative LDS byte offset
  return (unsigned int)(unsigned long long)p;
}
static __device__ __forceinline__ void async_copy_b128(unsigned int lds_off, const void* g) {
  unsigned long long ga = (unsigned long long)g;
  asm volatile("global_load_async_to_lds_b128 %0, %1, off"
               :: "v"(lds_off), "v"(ga) : "memory");
}
static __device__ __forceinline__ void wait_async_0() {
  asm volatile("s_wait_asynccnt 0x0" ::: "memory");
}

// ---------------- wave32 half-group reductions (lanes 0-15 / 16-31 separate)
static __device__ __forceinline__ float half_reduce_max(float v) {
#pragma unroll
  for (int off = 1; off < 16; off <<= 1) v = fmaxf(v, __shfl_xor(v, off, 32));
  return v;
}
static __device__ __forceinline__ float half_reduce_sum(float v) {
#pragma unroll
  for (int off = 1; off < 16; off <<= 1) v += __shfl_xor(v, off, 32);
  return v;
}

// A fragment, 16-bit 16x32: lanes 0-15 hold K {0..7,16..23}, lanes 16-31 K {8..15,24..31}.
static __device__ __forceinline__ v16h load_a_frag(const _Float16* p) {
  v16h a;
#pragma unroll
  for (int i = 0; i < 8; ++i) { a[i] = p[i]; a[8 + i] = p[16 + i]; }
  return a;
}
// B fragment, 16-bit 32x16: lanes 0-15 hold K {0..7,16..23}, lanes 16-31 K {8..15,24..31}.
static __device__ __forceinline__ v16h load_b_frag(const _Float16* p) {
  v16h b;
#pragma unroll
  for (int i = 0; i < 8; ++i) { b[i] = p[i]; b[8 + i] = p[16 + i]; }
  return b;
}

// ---------------- LayerNorm over a 768-wide row (f32 out + optional f16 copy)
static __device__ __forceinline__ void ln_row_768(float v0, float v1, float v2,
                                                  const float* g, const float* b,
                                                  float* outrow, _Float16* outrowh, int t) {
  __shared__ float red[256];
  red[t] = v0 + v1 + v2;
  __syncthreads();
#pragma unroll
  for (int o = 128; o > 0; o >>= 1) { if (t < o) red[t] += red[t + o]; __syncthreads(); }
  float mu = red[0] * (1.0f / 768.0f);
  __syncthreads();
  float d0 = v0 - mu, d1 = v1 - mu, d2 = v2 - mu;
  red[t] = d0 * d0 + d1 * d1 + d2 * d2;
  __syncthreads();
#pragma unroll
  for (int o = 128; o > 0; o >>= 1) { if (t < o) red[t] += red[t + o]; __syncthreads(); }
  float rstd = rsqrtf(red[0] * (1.0f / 768.0f) + 1e-5f);
  float y0 = d0 * rstd * g[t]       + b[t];
  float y1 = d1 * rstd * g[t + 256] + b[t + 256];
  float y2 = d2 * rstd * g[t + 512] + b[t + 512];
  outrow[t] = y0; outrow[t + 256] = y1; outrow[t + 512] = y2;
  if (outrowh) {
    outrowh[t] = (_Float16)y0; outrowh[t + 256] = (_Float16)y1; outrowh[t + 512] = (_Float16)y2;
  }
}

__global__ void embed_ln_kernel(const int* __restrict__ ids,
                                const float* __restrict__ wemb,
                                const float* __restrict__ pemb,
                                const float* __restrict__ g, const float* __restrict__ b,
                                float* __restrict__ X, _Float16* __restrict__ Xh) {
  int row = blockIdx.x, t = threadIdx.x;
  int id = ids[row];
  const float* wr = wemb + (size_t)id * D_MODEL;
  const float* pr = pemb + (size_t)(row + 2) * D_MODEL;
  float v0 = wr[t]       + pr[t];
  float v1 = wr[t + 256] + pr[t + 256];
  float v2 = wr[t + 512] + pr[t + 512];
  ln_row_768(v0, v1, v2, g, b, X + (size_t)row * D_MODEL,
             Xh + (size_t)row * D_MODEL, t);
}

__global__ void ln_kernel(const float* __restrict__ in,
                          const float* __restrict__ g, const float* __restrict__ b,
                          float* __restrict__ out, _Float16* __restrict__ outh) {
  int row = blockIdx.x, t = threadIdx.x;
  const float* r = in + (size_t)row * D_MODEL;
  ln_row_768(r[t], r[t + 256], r[t + 512], g, b,
             out + (size_t)row * D_MODEL, outh + (size_t)row * D_MODEL, t);
}

// ---------------------------------------------------------------------------
// Weight stage: Wt[n][k] = (f16) W[k][n].  Tiled 32x32 transpose through LDS.
// ---------------------------------------------------------------------------
__global__ void wconv_kernel(const float* __restrict__ W, _Float16* __restrict__ Wt,
                             int K, int N) {
  __shared__ float tile[32][33];
  int k0 = blockIdx.y * 32, n0 = blockIdx.x * 32;
  int tx = threadIdx.x, ty = threadIdx.y;  // (32, 8)
#pragma unroll
  for (int i = 0; i < 32; i += 8)
    tile[ty + i][tx] = W[(size_t)(k0 + ty + i) * N + n0 + tx];
  __syncthreads();
#pragma unroll
  for (int i = 0; i < 32; i += 8)
    Wt[(size_t)(n0 + ty + i) * K + k0 + tx] = (_Float16)tile[tx][ty + i];
}

// ---------------------------------------------------------------------------
// WMMA GEMM: C[M,N] = act(A[M,K] @ B[K,N] + bias [+ res]),
// A: f16 row-major [M][K];  Bt: f16 TRANSPOSED weights [N][K].
// Block = 256 thr (8 waves), block tile 128x64, BK=32.
// Wave grid 4(M) x 2(N); each wave computes 32x32 = 4 WMMAs per K-step.
// Double-buffered async->LDS with a single barrier per K-step:
//   [wait_async 0] [barrier] [prefetch s+1] [compute s]
// act: 0 none, 1 exact GELU. Cf (f32) / Ch (f16) optional outputs.
// ---------------------------------------------------------------------------
__global__ void gemm_wmma_kernel(const _Float16* __restrict__ A, const _Float16* __restrict__ Bt,
                                 const float* __restrict__ bias, const float* __restrict__ res,
                                 float* __restrict__ Cf, _Float16* __restrict__ Ch,
                                 int M, int N, int K, int act) {
  __shared__ _Float16 As[2][128][40];  // [buf][m][k], 8-half row pad (80B rows, 16B aligned)
  __shared__ _Float16 Bs[2][64][40];   // [buf][n][k]

  const int tid = threadIdx.x;
  const int lane = tid & 31, wave = tid >> 5;
  const int wm = wave >> 1, wn = wave & 1;          // 4 x 2 wave grid
  const int bm = blockIdx.y * 128, bn = blockIdx.x * 64;
  const int r = lane & 15;
  const bool hi = lane >= 16;

  v8f zero8 = {0, 0, 0, 0, 0, 0, 0, 0};
  v8f acc[2][2];
#pragma unroll
  for (int u = 0; u < 2; ++u)
#pragma unroll
    for (int v = 0; v < 2; ++v) acc[u][v] = zero8;

  // per K-step each thread async-copies: 2 x 16B of A, 1 x 16B of B
  const int ar0 = tid >> 2;                     // rows 0..63
  const int ar1 = 64 + (tid >> 2);              // rows 64..127
  const int akq = (tid & 3) * 8;                // 0,8,16,24 halves
  const _Float16* aG0 = A  + (size_t)(bm + ar0) * K + akq;
  const _Float16* aG1 = A  + (size_t)(bm + ar1) * K + akq;
  const _Float16* bG  = Bt + (size_t)(bn + ar0) * K + akq;
  const unsigned int aL0[2] = { lds_off_of(&As[0][ar0][akq]), lds_off_of(&As[1][ar0][akq]) };
  const unsigned int aL1[2] = { lds_off_of(&As[0][ar1][akq]), lds_off_of(&As[1][ar1][akq]) };
  const unsigned int bL[2]  = { lds_off_of(&Bs[0][ar0][akq]), lds_off_of(&Bs[1][ar0][akq]) };

  const int nsteps = K >> 5;

  // prologue: prefetch step 0 into buffer 0
  async_copy_b128(aL0[0], aG0);
  async_copy_b128(aL1[0], aG1);
  async_copy_b128(bL[0],  bG);

  for (int s = 0; s < nsteps - 1; ++s) {
    const int cur = s & 1, nb = cur ^ 1;
    wait_async_0();          // own async writes of tile s have landed
    __syncthreads();         // everyone's tile s visible; frees buffer nb (read in s-1)
    const int koff = (s + 1) * 32;
    async_copy_b128(aL0[nb], aG0 + koff);
    async_copy_b128(aL1[nb], aG1 + koff);
    async_copy_b128(bL[nb],  bG  + koff);

    v16h af0 = load_a_frag(&As[cur][wm * 32 + r][hi ? 8 : 0]);
    v16h af1 = load_a_frag(&As[cur][wm * 32 + 16 + r][hi ? 8 : 0]);
    v16h bf0 = load_b_frag(&Bs[cur][wn * 32 + r][hi ? 8 : 0]);
    v16h bf1 = load_b_frag(&Bs[cur][wn * 32 + 16 + r][hi ? 8 : 0]);
    acc[0][0] = __builtin_amdgcn_wmma_f32_16x16x32_f16(false, af0, false, bf0, (short)0, acc[0][0], false, false);
    acc[0][1] = __builtin_amdgcn_wmma_f32_16x16x32_f16(false, af0, false, bf1, (short)0, acc[0][1], false, false);
    acc[1][0] = __builtin_amdgcn_wmma_f32_16x16x32_f16(false, af1, false, bf0, (short)0, acc[1][0], false, false);
    acc[1][1] = __builtin_amdgcn_wmma_f32_16x16x32_f16(false, af1, false, bf1, (short)0, acc[1][1], false, false);
  }

  // last step (no prefetch)
  {
    const int cur = (nsteps - 1) & 1;
    wait_async_0();
    __syncthreads();
    v16h af0 = load_a_frag(&As[cur][wm * 32 + r][hi ? 8 : 0]);
    v16h af1 = load_a_frag(&As[cur][wm * 32 + 16 + r][hi ? 8 : 0]);
    v16h bf0 = load_b_frag(&Bs[cur][wn * 32 + r][hi ? 8 : 0]);
    v16h bf1 = load_b_frag(&Bs[cur][wn * 32 + 16 + r][hi ? 8 : 0]);
    acc[0][0] = __builtin_amdgcn_wmma_f32_16x16x32_f16(false, af0, false, bf0, (short)0, acc[0][0], false, false);
    acc[0][1] = __builtin_amdgcn_wmma_f32_16x16x32_f16(false, af0, false, bf1, (short)0, acc[0][1], false, false);
    acc[1][0] = __builtin_amdgcn_wmma_f32_16x16x32_f16(false, af1, false, bf0, (short)0, acc[1][0], false, false);
    acc[1][1] = __builtin_amdgcn_wmma_f32_16x16x32_f16(false, af1, false, bf1, (short)0, acc[1][1], false, false);
  }

  // epilogue: C layout -> VGPR i holds rows (i | i+8), col = lane&15
#pragma unroll
  for (int u = 0; u < 2; ++u) {
#pragma unroll
    for (int v = 0; v < 2; ++v) {
      int gc = bn + wn * 32 + v * 16 + r;
#pragma unroll
      for (int i = 0; i < 8; ++i) {
        int gr = bm + wm * 32 + u * 16 + i + (hi ? 8 : 0);
        float val = acc[u][v][i] + bias[gc];
        if (res) val += res[(size_t)gr * N + gc];
        if (act == 1) val = 0.5f * val * (1.0f + erff(val * 0.70710678118f));
        size_t o = (size_t)gr * N + gc;
        if (Cf) Cf[o] = val;
        if (Ch) Ch[o] = (_Float16)val;
      }
    }
  }
}

// ---------------------------------------------------------------------------
// Sliding-window flash attention. One workgroup (512 thr = 16 waves) per
// (chunk n, head h). K chunk lands in LDS via async copies; V is transposed
// into LDS; online softmax; WMMA for both einsums. Output written as f16.
// Dynamic LDS: Ks[256][72] + VsT[64][264] + Ps[16 waves][16][72].
// ---------------------------------------------------------------------------
__global__ void attn_kernel(const _Float16* __restrict__ qh, const _Float16* __restrict__ kh,
                            const _Float16* __restrict__ vh, const int* __restrict__ am,
                            _Float16* __restrict__ outh) {
  extern __shared__ _Float16 smem[];
  _Float16* Ks = smem;                  // [key][64+8]
  _Float16* Vs = Ks + 256 * 72;         // [dh][256+8]  (transposed V)
  _Float16* Ps = Vs + 64 * 264;         // [wave*16+row][64+8]

  const int n = blockIdx.x, h = blockIdx.y;
  const int tid = threadIdx.x;
  const int lane = tid & 31, wave = tid >> 5;
  const int r = lane & 15;
  const bool hi = lane >= 16;
  const int w0 = wave * 16;             // query-row base within chunk
  const int qbase = n * 256;

  // Q fragments for this wave's 16 rows (dh=64 -> two K-steps of 32)
  v16h qa[2];
  {
    const _Float16* qp = qh + (size_t)(qbase + w0 + r) * D_MODEL + h * DHEAD;
#pragma unroll
    for (int kk = 0; kk < 2; ++kk) {
      const _Float16* p = qp + kk * 32 + (hi ? 8 : 0);
#pragma unroll
      for (int i = 0; i < 8; ++i) { qa[kk][i] = p[i]; qa[kk][8 + i] = p[16 + i]; }
    }
  }

  v8f zero8 = {0, 0, 0, 0, 0, 0, 0, 0};
  v8f oacc[4];
#pragma unroll
  for (int t = 0; t < 4; ++t) oacc[t] = zero8;
  float m[8], l[8];
#pragma unroll
  for (int i = 0; i < 8; ++i) { m[i] = -3.0e38f; l[i] = 0.0f; }

  const int lkey = tid >> 1;            // 0..255
  const int ld0  = (tid & 1) * 32;      // dh half (32 halves = 64B per thread)

  for (int j = 0; j < 3; ++j) {
    const int ks = (n + j - 1) * 256;   // global index of first key in chunk
    // ---- fill Ks via async global->LDS (zero pad OOB), Vs transposed via ds
    {
      int gk = ks + lkey;
      bool ok = (gk >= 0) && (gk < S_LEN);
      unsigned int kd = lds_off_of(&Ks[lkey * 72 + ld0]);
      if (ok) {
        const _Float16* kp = kh + (size_t)gk * D_MODEL + h * DHEAD + ld0;
#pragma unroll
        for (int i = 0; i < 4; ++i)
          async_copy_b128(kd + 16u * i, kp + 8 * i);
        const _Float16* vp = vh + (size_t)gk * D_MODEL + h * DHEAD + ld0;
#pragma unroll
        for (int i = 0; i < 32; ++i) Vs[(ld0 + i) * 264 + lkey] = vp[i];
      } else {
        uint4 z; z.x = 0; z.y = 0; z.z = 0; z.w = 0;
        uint4* kz = (uint4*)(&Ks[lkey * 72 + ld0]);
#pragma unroll
        for (int i = 0; i < 4; ++i) kz[i] = z;
#pragma unroll
        for (int i = 0; i < 32; ++i) Vs[(ld0 + i) * 264 + lkey] = (_Float16)0.0f;
      }
    }
    wait_async_0();
    __syncthreads();

    // ---- 4 key sub-tiles of 64
#pragma unroll
    for (int c4 = 0; c4 < 4; ++c4) {
      const int c0 = c4 * 64;
      const int relmax = (j - 1) * 256 + c0 + 63 - w0;
      const int relmin = (j - 1) * 256 + c0 - (w0 + 15);
      if (relmax < -WWIN || relmin > WWIN) continue;   // band prune

      // scores S = Q(16x64) . K^T -> 4 tiles of 16x16
      v8f sf[4];
#pragma unroll
      for (int t = 0; t < 4; ++t) {
        sf[t] = zero8;
#pragma unroll
        for (int kk = 0; kk < 2; ++kk) {
          v16h bf = load_b_frag(&Ks[(c0 + t * 16 + r) * 72 + kk * 32 + (hi ? 8 : 0)]);
          sf[t] = __builtin_amdgcn_wmma_f32_16x16x32_f16(false, qa[kk], false, bf,
                                                         (short)0, sf[t], false, false);
        }
      }

      // scale + band/pad/key-mask, row max
      float sv[4][8];
      float rmax[8];
#pragma unroll
      for (int i = 0; i < 8; ++i) rmax[i] = -3.0e38f;
#pragma unroll
      for (int t = 0; t < 4; ++t) {
        int gk = ks + c0 + t * 16 + r;
        bool kin = (gk >= 0) && (gk < S_LEN);
        bool mok = false;
        if (kin) mok = (am[gk] != 0);
#pragma unroll
        for (int i = 0; i < 8; ++i) {
          int qr = w0 + i + (hi ? 8 : 0);
          int rel = gk - (qbase + qr);
          bool ok = mok && (rel >= -WWIN) && (rel <= WWIN);
          float v = ok ? sf[t][i] * 0.125f : NEGBIG;
          sv[t][i] = v;
          rmax[i] = fmaxf(rmax[i], v);
        }
      }
#pragma unroll
      for (int i = 0; i < 8; ++i) rmax[i] = half_reduce_max(rmax[i]);

      // online-softmax rescale
      float fac[8];
#pragma unroll
      for (int i = 0; i < 8; ++i) {
        float mn = fmaxf(m[i], rmax[i]);
        fac[i] = __expf(m[i] - mn);
        m[i] = mn;
        l[i] *= fac[i];
      }
#pragma unroll
      for (int t = 0; t < 4; ++t)
#pragma unroll
        for (int i = 0; i < 8; ++i) oacc[t][i] *= fac[i];

      // P = exp(S - m), stage to per-wave LDS strip (C layout -> A layout)
      float rsum[8];
#pragma unroll
      for (int i = 0; i < 8; ++i) rsum[i] = 0.0f;
#pragma unroll
      for (int t = 0; t < 4; ++t) {
#pragma unroll
        for (int i = 0; i < 8; ++i) {
          float p = __expf(sv[t][i] - m[i]);
          rsum[i] += p;
          Ps[(w0 + i + (hi ? 8 : 0)) * 72 + t * 16 + r] = (_Float16)p;
        }
      }
#pragma unroll
      for (int i = 0; i < 8; ++i) l[i] += half_reduce_sum(rsum[i]);

      asm volatile("" ::: "memory");   // LDS in-order per wave; fence the compiler

      // O += P(16x64) . V(64x64)
#pragma unroll
      for (int kk = 0; kk < 2; ++kk) {
        v16h pa = load_a_frag(&Ps[(w0 + r) * 72 + kk * 32 + (hi ? 8 : 0)]);
#pragma unroll
        for (int t = 0; t < 4; ++t) {
          v16h bv = load_b_frag(&Vs[(t * 16 + r) * 264 + c0 + kk * 32 + (hi ? 8 : 0)]);
          oacc[t] = __builtin_amdgcn_wmma_f32_16x16x32_f16(false, pa, false, bv,
                                                           (short)0, oacc[t], false, false);
        }
      }
    }
    __syncthreads();
  }

  // normalize + write O as f16 to [S][D] (head slice) for the O-projection GEMM
  float invl[8];
#pragma unroll
  for (int i = 0; i < 8; ++i) invl[i] = (l[i] > 0.0f) ? (1.0f / l[i]) : 0.0f;
#pragma unroll
  for (int t = 0; t < 4; ++t) {
    int d = h * DHEAD + t * 16 + r;
#pragma unroll
    for (int i = 0; i < 8; ++i) {
      int gr = qbase + w0 + i + (hi ? 8 : 0);
      outh[(size_t)gr * D_MODEL + d] = (_Float16)(oacc[t][i] * invl[i]);
    }
  }
}

// ---------------------------------------------------------------------------
__global__ void sep_kernel(const int* __restrict__ ids, int* __restrict__ sep) {
  __shared__ int red[256];
  int t = threadIdx.x, best = -1;
  for (int i = t; i < S_LEN; i += 256)
    if (ids[i] == 2) best = i;
  red[t] = best;
  __syncthreads();
#pragma unroll
  for (int o = 128; o > 0; o >>= 1) { if (t < o) red[t] = max(red[t], red[t + o]); __syncthreads(); }
  if (t == 0) sep[0] = (red[0] < 0) ? (S_LEN - 1) : red[0];
}

__global__ void head_kernel(const float* __restrict__ X, const int* __restrict__ sep,
                            const float* __restrict__ h1w, const float* __restrict__ h1b,
                            const float* __restrict__ h2w, const float* __restrict__ h2b,
                            const float* __restrict__ h3w, const float* __restrict__ h3b,
                            float* __restrict__ out) {
  __shared__ float emb[768];
  __shared__ float z1[512];
  __shared__ float z2[256];
  int t = threadIdx.x;  // 512 threads
  int srow = sep[0];
  for (int i = t; i < 768; i += 512) emb[i] = X[(size_t)srow * D_MODEL + i];
  __syncthreads();
  {
    float s = h1b[t];
    for (int k = 0; k < 768; ++k) s += emb[k] * h1w[k * 512 + t];
    z1[t] = fmaxf(s, 0.0f);
  }
  __syncthreads();
  if (t < 256) {
    float s = h2b[t];
    for (int k = 0; k < 512; ++k) s += z1[k] * h2w[k * 256 + t];
    z2[t] = fmaxf(s, 0.0f);
  }
  __syncthreads();
  if (t == 0) {
    float s = h3b[0];
    for (int k = 0; k < 256; ++k) s += z2[k] * h3w[k];
    out[0] = tanhf(s);
  }
}

// ---------------------------------------------------------------------------
extern "C" void kernel_launch(void* const* d_in, const int* in_sizes, int n_in,
                              void* d_out, int out_size, void* d_ws, size_t ws_size,
                              hipStream_t stream) {
  (void)in_sizes; (void)n_in; (void)out_size; (void)ws_size;
  const int*   ids  = (const int*)d_in[0];
  const int*   am   = (const int*)d_in[1];
  const float* wemb = (const float*)d_in[2];
  const float* pemb = (const float*)d_in[3];
  const float* elns = (const float*)d_in[4];
  const float* elnb = (const float*)d_in[5];
  const float* Wq = (const float*)d_in[6];  const float* bq = (const float*)d_in[7];
  const float* Wk = (const float*)d_in[8];  const float* bk = (const float*)d_in[9];
  const float* Wv = (const float*)d_in[10]; const float* bv = (const float*)d_in[11];
  const float* Wo = (const float*)d_in[12]; const float* bo = (const float*)d_in[13];
  const float* l1s = (const float*)d_in[14]; const float* l1b = (const float*)d_in[15];
  const float* W1 = (const float*)d_in[16]; const float* b1 = (const float*)d_in[17];
  const float* W2 = (const float*)d_in[18]; const float* b2 = (const float*)d_in[19];
  const float* l2s = (const float*)d_in[20]; const float* l2b = (const float*)d_in[21];
  const float* h1w = (const float*)d_in[22]; const float* h1b = (const float*)d_in[23];
  const float* h2w = (const float*)d_in[24]; const float* h2b = (const float*)d_in[25];
  const float* h3w = (const float*)d_in[26]; const float* h3b = (const float*)d_in[27];

  // ---- workspace layout
  float* X  = (float*)d_ws;                            // [S][D] f32
  float* T1 = X  + (size_t)S_LEN * D_MODEL;            // [S][D] f32
  _Float16* Xh  = (_Float16*)(T1 + (size_t)S_LEN * D_MODEL);   // [S][D] f16
  _Float16* AOh = Xh  + (size_t)S_LEN * D_MODEL;       // [S][D]  f16 (attention out)
  _Float16* FFh = AOh + (size_t)S_LEN * D_MODEL;       // [S][FF] f16
  _Float16* QH  = FFh + (size_t)S_LEN * FFDIM;
  _Float16* KH  = QH  + (size_t)S_LEN * D_MODEL;
  _Float16* VH  = KH  + (size_t)S_LEN * D_MODEL;
  // per-layer staged f16 transposed weights (reused each layer; stream-ordered)
  _Float16* WqT = VH  + (size_t)S_LEN * D_MODEL;       // [768][768]
  _Float16* WkT = WqT + (size_t)D_MODEL * D_MODEL;
  _Float16* WvT = WkT + (size_t)D_MODEL * D_MODEL;
  _Float16* WoT = WvT + (size_t)D_MODEL * D_MODEL;
  _Float16* W1T = WoT + (size_t)D_MODEL * D_MODEL;     // [3072][768]
  _Float16* W2T = W1T + (size_t)D_MODEL * FFDIM;       // [768][3072]
  int* SEP = (int*)(W2T + (size_t)FFDIM * D_MODEL);

  const size_t attn_shmem = (size_t)(256 * 72 + 64 * 264 + 256 * 72) * sizeof(_Float16);
  (void)hipFuncSetAttribute((const void*)attn_kernel,
                            hipFuncAttributeMaxDynamicSharedMemorySize, (int)attn_shmem);

  embed_ln_kernel<<<S_LEN, 256, 0, stream>>>(ids, wemb, pemb, elns, elnb, X, Xh);

  dim3 wb(32, 8);
  dim3 gwDD(D_MODEL / 32, D_MODEL / 32);
  dim3 gwDF(FFDIM / 32, D_MODEL / 32);
  dim3 gwFD(D_MODEL / 32, FFDIM / 32);
  dim3 gD(D_MODEL / 64, S_LEN / 128);
  dim3 gF(FFDIM / 64, S_LEN / 128);
  dim3 gA(S_LEN / 256, NHEAD);

  for (int l = 0; l < NLAYER; ++l) {
    const float* wq = Wq + (size_t)l * D_MODEL * D_MODEL; const float* bql = bq + (size_t)l * D_MODEL;
    const float* wk = Wk + (size_t)l * D_MODEL * D_MODEL; const float* bkl = bk + (size_t)l * D_MODEL;
    const float* wv = Wv + (size_t)l * D_MODEL * D_MODEL; const float* bvl = bv + (size_t)l * D_MODEL;
    const float* wo = Wo + (size_t)l * D_MODEL * D_MODEL; const float* bol = bo + (size_t)l * D_MODEL;
    const float* w1 = W1 + (size_t)l * D_MODEL * FFDIM;   const float* b1l = b1 + (size_t)l * FFDIM;
    const float* w2 = W2 + (size_t)l * FFDIM * D_MODEL;   const float* b2l = b2 + (size_t)l * D_MODEL;
    const float* l1sl = l1s + (size_t)l * D_MODEL; const float* l1bl = l1b + (size_t)l * D_MODEL;
    const float* l2sl = l2s + (size_t)l * D_MODEL; const float* l2bl = l2b + (size_t)l * D_MODEL;

    // stage this layer's weights as f16 [N][K]
    wconv_kernel<<<gwDD, wb, 0, stream>>>(wq, WqT, D_MODEL, D_MODEL);
    wconv_kernel<<<gwDD, wb, 0, stream>>>(wk, WkT, D_MODEL, D_MODEL);
    wconv_kernel<<<gwDD, wb, 0, stream>>>(wv, WvT, D_MODEL, D_MODEL);
    wconv_kernel<<<gwDD, wb, 0, stream>>>(wo, WoT, D_MODEL, D_MODEL);
    wconv_kernel<<<gwDF, wb, 0, stream>>>(w1, W1T, D_MODEL, FFDIM);
    wconv_kernel<<<gwFD, wb, 0, stream>>>(w2, W2T, FFDIM, D_MODEL);

    // Q,K,V projections -> f16
    gemm_wmma_kernel<<<gD, 256, 0, stream>>>(Xh, WqT, bql, nullptr, nullptr, QH,
                                             S_LEN, D_MODEL, D_MODEL, 0);
    gemm_wmma_kernel<<<gD, 256, 0, stream>>>(Xh, WkT, bkl, nullptr, nullptr, KH,
                                             S_LEN, D_MODEL, D_MODEL, 0);
    gemm_wmma_kernel<<<gD, 256, 0, stream>>>(Xh, WvT, bvl, nullptr, nullptr, VH,
                                             S_LEN, D_MODEL, D_MODEL, 0);
    // sliding-window flash attention
    attn_kernel<<<gA, 512, attn_shmem, stream>>>(QH, KH, VH, am, AOh);
    // output projection + residual, then LN1
    gemm_wmma_kernel<<<gD, 256, 0, stream>>>(AOh, WoT, bol, X, T1, nullptr,
                                             S_LEN, D_MODEL, D_MODEL, 0);
    ln_kernel<<<S_LEN, 256, 0, stream>>>(T1, l1sl, l1bl, X, Xh);
    // FFN
    gemm_wmma_kernel<<<gF, 256, 0, stream>>>(Xh, W1T, b1l, nullptr, nullptr, FFh,
                                             S_LEN, FFDIM, D_MODEL, 1);
    gemm_wmma_kernel<<<gD, 256, 0, stream>>>(FFh, W2T, b2l, X, T1, nullptr,
                                             S_LEN, D_MODEL, FFDIM, 0);
    ln_kernel<<<S_LEN, 256, 0, stream>>>(T1, l2sl, l2bl, X, Xh);
  }

  sep_kernel<<<1, 256, 0, stream>>>(ids, SEP);
  head_kernel<<<1, 512, 0, stream>>>(X, SEP, h1w, h1b, h2w, h2b, h3w, h3b, (float*)d_out);
}
